// AgentGNN_22265110463046
// MI455X (gfx1250) — matrix operand order repaired and checked by reading, not compiled, not run on hardware
//
#include <hip/hip_runtime.h>
#include <hip/hip_bf16.h>
#include <math.h>

// ---------------------------------------------------------------------------
// GATv2 x5 + GlobalAttention pooling + FF head for MI455X (gfx1250, wave32).
// Dense [N x K] @ [K x 32] linears run on v_wmma_f32_16x16x32_f16 (f16 in,
// f32 accumulate) with branchless float4 A-loads, pre-packed f16 B fragments,
// and unguarded immediate-offset stores into 16-row-padded buffers.
// Edge work is wave-per-edge (lane == channel): 128B-row gathers/scatters.
// ---------------------------------------------------------------------------

typedef __attribute__((ext_vector_type(16))) _Float16 v16h;
typedef __attribute__((ext_vector_type(8)))  float    v8f;

#define C_HID 32
#define NEG_SLOPE 0.2f

// ---------------- generic fill -------------------------------------------
__global__ __launch_bounds__(256) void fill_kernel(float* __restrict__ p, int n, float v) {
    int i = blockIdx.x * blockDim.x + threadIdx.x;
    if (i < n) p[i] = v;
}

// ---------------- float atomic max (sign-split trick) --------------------
__device__ inline void atomicMaxF(float* addr, float val) {
    if (val >= 0.f) atomicMax((int*)addr, __float_as_int(val));
    else            atomicMin((unsigned int*)addr, __float_as_uint(val));
}

// ---------------- self-loop edge_attr = mean of incoming -----------------
__global__ __launch_bounds__(256) void edge_count_kernel(
    const int* __restrict__ dst, const float* __restrict__ edge_attr,
    float* __restrict__ cnt, float* __restrict__ loop_sum, int E) {
    int e = blockIdx.x * blockDim.x + threadIdx.x;
    if (e >= E) return;
    int d = dst[e];
    atomicAdd(&cnt[d], 1.0f);
    atomicAdd(&loop_sum[d * 3 + 0], edge_attr[e * 3 + 0]);
    atomicAdd(&loop_sum[d * 3 + 1], edge_attr[e * 3 + 1]);
    atomicAdd(&loop_sum[d * 3 + 2], edge_attr[e * 3 + 2]);
}

__global__ __launch_bounds__(256) void loop_attr_kernel(
    const float* __restrict__ loop_sum, const float* __restrict__ cnt,
    float* __restrict__ loop_attr, int N) {
    int i = blockIdx.x * blockDim.x + threadIdx.x;
    if (i >= N * 3) return;
    float c = cnt[i / 3];
    loop_attr[i] = loop_sum[i] / fmaxf(c, 1.0f);
}

// ---------------- pack W[Kx32] (f32, row-major) into B-fragment f16 ------
// Wp layout: [tile 0..1][lane 0..31][16 halves]; lane half h holds
// K=[8h..8h+7] then [16+8h..16+8h+7] (2 per VGPR), zero-padded for K<32.
__global__ __launch_bounds__(64) void pack_w_kernel(
    const float* __restrict__ W, int K, _Float16* __restrict__ Wp) {
    int tid  = threadIdx.x;            // 0..63
    int t    = tid >> 5;
    int lane = tid & 31;
    int half = lane >> 4, nloc = lane & 15;
    int c = t * 16 + nloc;
#pragma unroll
    for (int v = 0; v < 8; ++v) {
        int kb = (v < 4) ? (2 * v + 8 * half) : (16 + 2 * (v - 4) + 8 * half);
        float w0 = (kb     < K) ? W[(size_t)kb * C_HID + c]       : 0.f;
        float w1 = (kb + 1 < K) ? W[(size_t)(kb + 1) * C_HID + c] : 0.f;
        Wp[((size_t)t * 32 + lane) * 16 + 2 * v]     = (_Float16)w0;
        Wp[((size_t)t * 32 + lane) * 16 + 2 * v + 1] = (_Float16)w1;
    }
}

// ---------------- WMMA dual linear ---------------------------------------
// Y0[Mx32] = X[MxK] @ W0, Y1[Mx32] = X[MxK] @ W1 (Wp1 may be null).
// One wave -> 16 rows x 32 cols per output (two 16x16 WMMA tiles each).
// Y buffers MUST be padded to a multiple of 16 rows (stores are unguarded).
__global__ __launch_bounds__(256) void wmma_linear_kernel(
    const float* __restrict__ X, int M, int K,
    const _Float16* __restrict__ Wp0, const float* __restrict__ b0,
    float* __restrict__ Y0, int relu0,
    const _Float16* __restrict__ Wp1, const float* __restrict__ b1,
    float* __restrict__ Y1, int relu1) {
    const int lane = threadIdx.x & 31;
    const int wave = blockIdx.x * (blockDim.x >> 5) + (threadIdx.x >> 5);
    const int row0 = wave * 16;
    if (row0 >= M) return;                  // wave-uniform: EXEC all-1 at WMMA

    const int half = lane >> 4;             // 0: lanes 0-15, 1: lanes 16-31
    const int nloc = lane & 15;
    const int m    = row0 + nloc;           // A-row this lane supplies
    const int mc   = (m < M) ? m : (M - 1); // clamp: load address always valid

    // A fragment: lane holds K = [8h..8h+7] then [16+8h..16+8h+7] of row mc.
    v16h a;
    if (K == C_HID) {
        const float* rowp = X + (size_t)mc * C_HID;
        float4 c0 = *(const float4*)(rowp + 8 * half);
        float4 c1 = *(const float4*)(rowp + 8 * half + 4);
        float4 c2 = *(const float4*)(rowp + 16 + 8 * half);
        float4 c3 = *(const float4*)(rowp + 16 + 8 * half + 4);
        a[0]  = (_Float16)c0.x; a[1]  = (_Float16)c0.y;
        a[2]  = (_Float16)c0.z; a[3]  = (_Float16)c0.w;
        a[4]  = (_Float16)c1.x; a[5]  = (_Float16)c1.y;
        a[6]  = (_Float16)c1.z; a[7]  = (_Float16)c1.w;
        a[8]  = (_Float16)c2.x; a[9]  = (_Float16)c2.y;
        a[10] = (_Float16)c2.z; a[11] = (_Float16)c2.w;
        a[12] = (_Float16)c3.x; a[13] = (_Float16)c3.y;
        a[14] = (_Float16)c3.z; a[15] = (_Float16)c3.w;
    } else {
        const float* rowp = X + (size_t)mc * K;
#pragma unroll
        for (int v = 0; v < 8; ++v) {
            int kb = (v < 4) ? (2 * v + 8 * half) : (16 + 2 * (v - 4) + 8 * half);
            int k0 = (kb     < K) ? kb     : 0;      // clamped: branchless load
            int k1 = (kb + 1 < K) ? kb + 1 : 0;
            float x0 = rowp[k0];
            float x1 = rowp[k1];
            a[2 * v]     = (_Float16)((kb     < K) ? x0 : 0.f);
            a[2 * v + 1] = (_Float16)((kb + 1 < K) ? x1 : 0.f);
        }
    }

#pragma unroll
    for (int j = 0; j < 2; ++j) {
        const _Float16* Wp = j ? Wp1 : Wp0;
        if (!Wp) break;                      // wave-uniform scalar branch
        const float* bias = j ? b1 : b0;
        float* Y          = j ? Y1 : Y0;
        const float lo    = (j ? relu1 : relu0) ? 0.f : -__builtin_inff();
#pragma unroll
        for (int t = 0; t < 2; ++t) {
            v16h bf = *(const v16h*)(Wp + ((size_t)t * 32 + lane) * 16);
            v8f acc = {};
            acc = __builtin_amdgcn_wmma_f32_16x16x32_f16(
                      false, a, false, bf, (short)0, acc, false, false);
            const int c    = t * 16 + nloc;  // D column for this lane
            const float bb = bias ? bias[c] : 0.f;
            // D rows for this lane: row0 + 8*half + r, r = 0..7.
            float* yp = Y + (size_t)(row0 + 8 * half) * C_HID + c;
#pragma unroll
            for (int r = 0; r < 8; ++r) {
                yp[(size_t)r * C_HID] = fmaxf(acc[r] + bb, lo);
            }
        }
    }
}

// ---------------- pass 1: logits + segment max over dst ------------------
// wave per edge; lane = channel. Edges [0,E) are real, [E,E+N) are self-loops.
__global__ __launch_bounds__(256) void edge_logits_kernel(
    const int* __restrict__ src, const int* __restrict__ dst,
    const float* __restrict__ edge_attr, const float* __restrict__ loop_attr,
    const float* __restrict__ xl, const float* __restrict__ xr,
    const float* __restrict__ We /*3x32*/, const float* __restrict__ att,
    float* __restrict__ logits, float* __restrict__ mlog, int E, int N) {
    const int lane = threadIdx.x & 31;
    const int wid  = blockIdx.x * (blockDim.x >> 5) + (threadIdx.x >> 5);
    if (wid >= E + N) return;
    int s, d; const float* eap;
    if (wid < E) {
        s = src[wid]; d = dst[wid];
        eap = edge_attr + (size_t)wid * 3;
    } else {
        s = d = wid - E;
        eap = loop_attr + (size_t)s * 3;
    }
    float ea0 = eap[0], ea1 = eap[1], ea2 = eap[2];
    float t = xl[(size_t)s * C_HID + lane] + xr[(size_t)d * C_HID + lane]
            + ea0 * We[lane] + ea1 * We[C_HID + lane] + ea2 * We[2 * C_HID + lane];
    t = (t > 0.f) ? t : NEG_SLOPE * t;                 // leaky_relu
    float p = t * att[lane];
#pragma unroll
    for (int off = 16; off > 0; off >>= 1) p += __shfl_xor(p, off, 32);
    if (lane == 0) {
        logits[wid] = p;
        atomicMaxF(&mlog[d], p);
    }
}

// ---------------- pass 2: unnormalized softmax accumulate ----------------
__global__ __launch_bounds__(256) void edge_accum_kernel(
    const int* __restrict__ src, const int* __restrict__ dst,
    const float* __restrict__ xl, const float* __restrict__ logits,
    const float* __restrict__ mlog, float* __restrict__ accum,
    float* __restrict__ denom, int E, int N) {
    const int lane = threadIdx.x & 31;
    const int wid  = blockIdx.x * (blockDim.x >> 5) + (threadIdx.x >> 5);
    if (wid >= E + N) return;
    int s, d;
    if (wid < E) { s = src[wid]; d = dst[wid]; }
    else         { s = d = wid - E; }
    float a = __expf(logits[wid] - mlog[d]);
    atomicAdd(&accum[(size_t)d * C_HID + lane], a * xl[(size_t)s * C_HID + lane]);
    if (lane == 0) atomicAdd(&denom[d], a);
}

// ---------------- per-node finalize: h = relu(accum/denom + bias) --------
__global__ __launch_bounds__(256) void node_finalize_kernel(
    const float* __restrict__ accum, const float* __restrict__ denom,
    const float* __restrict__ bias, float* __restrict__ h, int N) {
    int i = blockIdx.x * blockDim.x + threadIdx.x;
    if (i >= N * C_HID) return;
    int node = i >> 5, c = i & 31;
    float v = accum[i] / denom[node] + bias[c];
    h[i] = fmaxf(v, 0.f);
}

// ---------------- 32 -> 1 projection (gate head, no relu) ----------------
__global__ __launch_bounds__(256) void rowdot32_kernel(
    const float* __restrict__ X, const float* __restrict__ w,
    const float* __restrict__ b, float* __restrict__ y, int n) {
    int i = blockIdx.x * blockDim.x + threadIdx.x;
    if (i >= n) return;
    float s = b[0];
#pragma unroll
    for (int k = 0; k < C_HID; ++k) s += X[(size_t)i * C_HID + k] * w[k];
    y[i] = s;
}

// ---------------- graph softmax pooling ----------------------------------
__global__ __launch_bounds__(256) void gate_max_kernel(
    const float* __restrict__ gate, const int* __restrict__ batch,
    float* __restrict__ gm, int N) {
    int i = blockIdx.x * blockDim.x + threadIdx.x;
    if (i >= N) return;
    atomicMaxF(&gm[batch[i]], gate[i]);
}

__global__ __launch_bounds__(256) void pool_accum_kernel(
    const float* __restrict__ gate, const float* __restrict__ gm,
    const int* __restrict__ batch, const float* __restrict__ feat,
    float* __restrict__ pooled, float* __restrict__ gd, int N) {
    const int lane = threadIdx.x & 31;
    const int i    = blockIdx.x * (blockDim.x >> 5) + (threadIdx.x >> 5);
    if (i >= N) return;
    int b = batch[i];
    float g = __expf(gate[i] - gm[b]);
    atomicAdd(&pooled[(size_t)b * C_HID + lane], g * feat[(size_t)i * C_HID + lane]);
    if (lane == 0) atomicAdd(&gd[b], g);
}

// ---------------- final FF head over G graphs -----------------------------
__global__ __launch_bounds__(64) void final_ff_kernel(
    const float* __restrict__ pooled, const float* __restrict__ gd,
    const float* __restrict__ W0, const float* __restrict__ b0,
    const float* __restrict__ W1, const float* __restrict__ b1,
    const float* __restrict__ W2, const float* __restrict__ b2,
    float* __restrict__ out, int G) {
    int g = blockIdx.x * blockDim.x + threadIdx.x;
    if (g >= G) return;
    float v[C_HID], h1[C_HID];
    float invd = 1.f / gd[g];
    for (int c = 0; c < C_HID; ++c) v[c] = pooled[(size_t)g * C_HID + c] * invd;
    for (int o = 0; o < C_HID; ++o) {
        float s = b0[o];
        for (int k = 0; k < C_HID; ++k) s += v[k] * W0[k * C_HID + o];
        h1[o] = fmaxf(s, 0.f);
    }
    for (int o = 0; o < C_HID; ++o) {
        float s = b1[o];
        for (int k = 0; k < C_HID; ++k) s += h1[k] * W1[k * C_HID + o];
        v[o] = fmaxf(s, 0.f);
    }
    float s = b2[0];
    for (int k = 0; k < C_HID; ++k) s += v[k] * W2[k];
    out[g] = s;
}

// ===========================================================================
extern "C" void kernel_launch(void* const* d_in, const int* in_sizes, int n_in,
                              void* d_out, int out_size, void* d_ws, size_t ws_size,
                              hipStream_t stream) {
    // ---- input mapping --------------------------------------------------
    // Top level in setup_inputs() insertion order; params pytree flattened
    // with sorted dict keys at each level (jax convention):
    //   params -> ff(3 lins: W,b each), gat(5 layers: att,bias,lin_e.W,
    //             lin_l.W,lin_l.b,lin_r.W,lin_r.b), gate(3 lins), pool(2 lins)
    const float* x          = (const float*)d_in[0];
    const int*   edge_index = (const int*)  d_in[1];
    const float* edge_attr  = (const float*)d_in[2];
    const int*   batch      = (const int*)  d_in[3];

    const int N    = in_sizes[3];
    const int E    = in_sizes[2] / 3;
    const int C_IN = in_sizes[0] / N;
    const int G    = out_size;
    const int N16  = ((N + 15) / 16) * 16;  // padded rows for GEMM outputs
    const int* src = edge_index;
    const int* dst = edge_index + E;

    const float* ffW0 = (const float*)d_in[4];
    const float* ffb0 = (const float*)d_in[5];
    const float* ffW1 = (const float*)d_in[6];
    const float* ffb1 = (const float*)d_in[7];
    const float* ffW2 = (const float*)d_in[8];
    const float* ffb2 = (const float*)d_in[9];
    // gat layer l at base 10 + 7*l: att,bias,We,Wl,bl,Wr,br
    const float* gW0  = (const float*)d_in[45];
    const float* gb0  = (const float*)d_in[46];
    const float* gW1  = (const float*)d_in[47];
    const float* gb1  = (const float*)d_in[48];
    const float* gW2  = (const float*)d_in[49];
    const float* gb2  = (const float*)d_in[50];
    const float* pW0  = (const float*)d_in[51];
    const float* pb0  = (const float*)d_in[52];
    const float* pW1  = (const float*)d_in[53];
    const float* pb1  = (const float*)d_in[54];

    // ---- workspace carve-up (floats; GEMM outputs padded to 16 rows) ----
    float* ws = (float*)d_ws;
    size_t o = 0;
    _Float16* wpA    = (_Float16*)(ws + o); o += 512;   // 2*32*16 f16 = 2KB
    _Float16* wpB    = (_Float16*)(ws + o); o += 512;
    float* xl        = ws + o; o += (size_t)N16 * C_HID;
    float* xr        = ws + o; o += (size_t)N16 * C_HID;
    float* h         = ws + o; o += (size_t)N16 * C_HID;
    float* accum     = ws + o; o += (size_t)N16 * C_HID;
    float* denom     = ws + o; o += N;
    float* mlog      = ws + o; o += N;
    float* logits    = ws + o; o += (size_t)(E + N);
    float* loop_attr = ws + o; o += (size_t)N * 3;
    float* cnt       = ws + o; o += N;
    float* loop_sum  = ws + o; o += (size_t)N * 3;
    float* gateval   = ws + o; o += N;
    float* gm        = ws + o; o += G;
    float* gd        = ws + o; o += G;
    float* pooled    = ws + o; o += (size_t)G * C_HID;
    (void)ws_size; (void)n_in;

    const float NEG_INF = -__builtin_inff();
    const int TB = 256;
    auto blocks  = [](long long n, int tb) { return (unsigned)((n + tb - 1) / tb); };
    const int WPB = TB / 32;                          // waves per 256-thread block
    const unsigned edgeBlocks = (unsigned)(((long long)(E + N) + WPB - 1) / WPB);
    const unsigned gemmBlocks = (unsigned)(((long long)N + 16 * WPB - 1) / (16 * WPB));

    // ---- self-loop edge_attr (layer-invariant) --------------------------
    fill_kernel<<<blocks(N, TB), TB, 0, stream>>>(cnt, N, 0.f);
    fill_kernel<<<blocks((long long)N * 3, TB), TB, 0, stream>>>(loop_sum, N * 3, 0.f);
    edge_count_kernel<<<blocks(E, TB), TB, 0, stream>>>(dst, edge_attr, cnt, loop_sum, E);
    loop_attr_kernel<<<blocks((long long)N * 3, TB), TB, 0, stream>>>(loop_sum, cnt, loop_attr, N);

    // ---- 5 GATv2 layers --------------------------------------------------
    const float* hin = x;
    int K = C_IN;
    for (int l = 0; l < 5; ++l) {
        const int b0i = 10 + 7 * l;
        const float* att  = (const float*)d_in[b0i + 0];
        const float* gbias= (const float*)d_in[b0i + 1];
        const float* We   = (const float*)d_in[b0i + 2];
        const float* Wl   = (const float*)d_in[b0i + 3];
        const float* bl   = (const float*)d_in[b0i + 4];
        const float* Wr   = (const float*)d_in[b0i + 5];
        const float* br   = (const float*)d_in[b0i + 6];

        pack_w_kernel<<<1, 64, 0, stream>>>(Wl, K, wpA);
        pack_w_kernel<<<1, 64, 0, stream>>>(Wr, K, wpB);
        wmma_linear_kernel<<<gemmBlocks, TB, 0, stream>>>(
            hin, N, K, wpA, bl, xl, 0, wpB, br, xr, 0);

        fill_kernel<<<blocks((long long)N * C_HID, TB), TB, 0, stream>>>(accum, N * C_HID, 0.f);
        fill_kernel<<<blocks(N, TB), TB, 0, stream>>>(denom, N, 0.f);
        fill_kernel<<<blocks(N, TB), TB, 0, stream>>>(mlog, N, NEG_INF);

        edge_logits_kernel<<<edgeBlocks, TB, 0, stream>>>(
            src, dst, edge_attr, loop_attr, xl, xr, We, att, logits, mlog, E, N);
        edge_accum_kernel<<<edgeBlocks, TB, 0, stream>>>(
            src, dst, xl, logits, mlog, accum, denom, E, N);
        node_finalize_kernel<<<blocks((long long)N * C_HID, TB), TB, 0, stream>>>(
            accum, denom, gbias, h, N);

        hin = h;
        K = C_HID;
    }

    // ---- readout: gate MLP (relu,relu,->1) and pool MLP (relu,relu) -----
    // stage 1 (shared input h): gate1 -> xl, pool1 -> accum
    pack_w_kernel<<<1, 64, 0, stream>>>(gW0, C_HID, wpA);
    pack_w_kernel<<<1, 64, 0, stream>>>(pW0, C_HID, wpB);
    wmma_linear_kernel<<<gemmBlocks, TB, 0, stream>>>(
        h, N, C_HID, wpA, gb0, xl, 1, wpB, pb0, accum, 1);
    // stage 2: gate2(xl) -> xr ; pool2(accum) -> h (= feat)
    pack_w_kernel<<<1, 64, 0, stream>>>(gW1, C_HID, wpA);
    pack_w_kernel<<<1, 64, 0, stream>>>(pW1, C_HID, wpB);
    wmma_linear_kernel<<<gemmBlocks, TB, 0, stream>>>(
        xl, N, C_HID, wpA, gb1, xr, 1, (const _Float16*)nullptr, nullptr, nullptr, 0);
    wmma_linear_kernel<<<gemmBlocks, TB, 0, stream>>>(
        accum, N, C_HID, wpB, pb1, h, 1, (const _Float16*)nullptr, nullptr, nullptr, 0);
    rowdot32_kernel<<<blocks(N, TB), TB, 0, stream>>>(xr, gW2, gb2, gateval, N);

    // graph-wise softmax pooling (feat = h)
    fill_kernel<<<blocks(G, 64), 64, 0, stream>>>(gm, G, NEG_INF);
    fill_kernel<<<blocks(G, 64), 64, 0, stream>>>(gd, G, 0.f);
    fill_kernel<<<blocks((long long)G * C_HID, TB), TB, 0, stream>>>(pooled, G * C_HID, 0.f);
    gate_max_kernel<<<blocks(N, TB), TB, 0, stream>>>(gateval, batch, gm, N);
    pool_accum_kernel<<<blocks((long long)N * 32, TB), TB, 0, stream>>>(
        gateval, gm, batch, h, pooled, gd, N);

    // final FF head -> [G] critic values
    final_ff_kernel<<<blocks(G, 64), 64, 0, stream>>>(
        pooled, gd, ffW0, ffb0, ffW1, ffb1, ffW2, ffb2, (float*)d_out, G);
}
